// refine_8_87101936763084
// MI455X (gfx1250) — compile-verified, matrix operand order checked by reading, not run on hardware
//
// SnowflakeNet-style refinement forward for MI455X (gfx1250).
// All 1x1-conv GEMMs run through v_wmma_f32_16x16x32_f16 (f16 in / f32 acc).
// Weights are pre-converted once into a zero-padded f16 arena so the WMMA
// K-loop has branch-free global_load_b128 A-operands; activations stage
// through LDS with fused f32->f16 conversion. Epilogues fuse bias+BN+act+res.
#include <hip/hip_runtime.h>

typedef __attribute__((ext_vector_type(16))) _Float16 v16h;
typedef __attribute__((ext_vector_type(8)))  _Float16 v8h;
typedef __attribute__((ext_vector_type(8)))  float    v8f;

#define ACT_NONE 0
#define ACT_RELU 1
#define ACT_TANH 2

#define GSL(i, total) for (size_t i = (size_t)blockIdx.x * blockDim.x + threadIdx.x; \
                           i < (total); i += (size_t)gridDim.x * blockDim.x)

// ---------------------------------------------------------------------------
// Weight prep: W [O,C] f32 -> Wh [Opad64, Cpad32] f16, zero padded.
// ---------------------------------------------------------------------------
__global__ void w2h_kernel(const float* __restrict__ W, _Float16* __restrict__ Wh,
                           int O, int C, int Op, int Cp)
{
  size_t total = (size_t)Op * Cp;
  GSL(i, total) {
    int c = (int)(i % Cp); int o = (int)(i / Cp);
    Wh[i] = (o < O && c < C) ? (_Float16)W[(size_t)o * C + c] : (_Float16)0.f;
  }
}

// ---------------------------------------------------------------------------
// WMMA GEMM: Y[b,o,n] = act( g[o]*(sum_c W[o,c]*X[b,c,n] + bias[o]) + bet[o] + res )
// X: [B,C,N] f32, Wh: [*,Cpad] f16 padded, Y: [B,O,N].  N multiple of 16.
// 128 threads = 4 waves; each wave owns a 16x16 output tile.
// ---------------------------------------------------------------------------
__global__ __launch_bounds__(128) void gemm_wmma_kernel(
    const float* __restrict__ X, const _Float16* __restrict__ Wh,
    const float* __restrict__ bias, const float* __restrict__ gam,
    const float* __restrict__ bet, const float* __restrict__ res,
    float* __restrict__ Y, int C, int Cpad, int O, int N, int act)
{
  const int b    = blockIdx.z;
  const int n0   = blockIdx.x << 4;
  const int wv   = threadIdx.x >> 5;
  const int lane = threadIdx.x & 31;
  const int m0   = (blockIdx.y * 4 + wv) << 4;

  __shared__ __align__(16) _Float16 sB[16][40];  // [n][k]; 80B row stride

  const float* Xb = X + (size_t)b * C * N;
  v8f acc = {};

  const int kk = threadIdx.x >> 2;        // 0..31 : k row staged by this thread
  const int nn = (threadIdx.x & 3) << 2;  // 0,4,8,12 : first n column staged
  const int nL = lane & 15;
  const int kb = (lane < 16) ? 0 : 8;     // documented 16-bit A/B lane K split
  const _Float16* wrow = Wh + (size_t)(m0 + nL) * Cpad;  // rows zero-padded

  for (int k0 = 0; k0 < C; k0 += 32) {
    float x0 = 0.f, x1 = 0.f, x2 = 0.f, x3 = 0.f;
    if (k0 + kk < C) {
      const float* p = Xb + (size_t)(k0 + kk) * N + n0 + nn;
      x0 = p[0]; x1 = p[1]; x2 = p[2]; x3 = p[3];
    }
    __syncthreads();
    sB[nn + 0][kk] = (_Float16)x0;
    sB[nn + 1][kk] = (_Float16)x1;
    sB[nn + 2][kk] = (_Float16)x2;
    sB[nn + 3][kk] = (_Float16)x3;
    __syncthreads();

    // B fragment: two aligned 16B LDS reads per lane.
    v8h blo = *(const v8h*)&sB[nL][kb];
    v8h bhi = *(const v8h*)&sB[nL][16 + kb];
    v16h bf = __builtin_shufflevector(blo, bhi, 0, 1, 2, 3, 4, 5, 6, 7,
                                      8, 9, 10, 11, 12, 13, 14, 15);
    // A fragment: two branch-free global_load_b128 from the padded f16 arena.
    v8h alo = *(const v8h*)(wrow + k0 + kb);
    v8h ahi = *(const v8h*)(wrow + k0 + 16 + kb);
    v16h af = __builtin_shufflevector(alo, ahi, 0, 1, 2, 3, 4, 5, 6, 7,
                                      8, 9, 10, 11, 12, 13, 14, 15);

    acc = __builtin_amdgcn_wmma_f32_16x16x32_f16(false, af, false, bf,
                                                 (short)0, acc, false, false);
  }

  const int mh = (lane < 16) ? 0 : 8;
#pragma unroll
  for (int r = 0; r < 8; ++r) {
    int o = m0 + mh + r;
    if (o < O) {
      float v  = acc[r];
      float sc = gam ? gam[o] : 1.f;
      v *= sc;
      if (bias) v += sc * bias[o];
      if (bet)  v += bet[o];
      size_t yi = ((size_t)b * O + o) * N + n0 + nL;
      if (res) v += res[yi];
      if (act == ACT_RELU)      v = v > 0.f ? v : 0.f;
      else if (act == ACT_TANH) v = tanhf(v);
      Y[yi] = v;
    }
  }
}

// ---------------------------------------------------------------------------
// Farthest point sampling: one block per batch; xyz [B,3,N] -> idx [B,npoint]
// ---------------------------------------------------------------------------
__global__ __launch_bounds__(256) void fps_kernel(const float* __restrict__ xyz,
                                                  int N, int npoint, int* __restrict__ out)
{
  __shared__ float dist[2048];
  __shared__ float sd[256];
  __shared__ int   si[256];
  __shared__ float cpt[3];
  const int b = blockIdx.x, t = threadIdx.x;
  const float* X = xyz + (size_t)b * 3 * N;
  for (int i = t; i < N; i += 256) dist[i] = 1e10f;
  int far = 0;
  for (int it = 0; it < npoint; ++it) {
    if (t == 0) {
      out[(size_t)b * npoint + it] = far;
      cpt[0] = X[far]; cpt[1] = X[N + far]; cpt[2] = X[2 * N + far];
    }
    __syncthreads();
    float bm = -1.f; int bi = 0;
    for (int i = t; i < N; i += 256) {
      float dx = X[i] - cpt[0], dy = X[N + i] - cpt[1], dz = X[2 * N + i] - cpt[2];
      float d  = fminf(dist[i], dx * dx + dy * dy + dz * dz);
      dist[i] = d;
      if (d > bm) { bm = d; bi = i; }
    }
    sd[t] = bm; si[t] = bi;
    __syncthreads();
    for (int s = 128; s > 0; s >>= 1) {
      if (t < s && sd[t + s] > sd[t]) { sd[t] = sd[t + s]; si[t] = si[t + s]; }
      __syncthreads();
    }
    far = si[0];
    __syncthreads();
  }
}

__global__ void gatherp_kernel(const float* __restrict__ src, const int* __restrict__ idx,
                               float* __restrict__ out, int B, int C, int N, int S)
{
  size_t total = (size_t)B * C * S;
  GSL(i, total) {
    int s = (int)(i % S); size_t r = i / S;
    int c = (int)(r % C); int b = (int)(r / C);
    out[i] = src[((size_t)b * C + c) * N + idx[(size_t)b * S + s]];
  }
}

__global__ void qball_kernel(const float* __restrict__ xyz, const float* __restrict__ nxyz,
                             int* __restrict__ gi, float r2, int B, int N, int S, int K)
{
  GSL(i, (size_t)B * S) {
    int b = (int)(i / S), s = (int)(i % S);
    const float* X = xyz + (size_t)b * 3 * N;
    const float* Q = nxyz + (size_t)b * 3 * S;
    float qx = Q[s], qy = Q[S + s], qz = Q[2 * S + s];
    int* o = gi + ((size_t)b * S + s) * K;
    int cnt = 0;
    for (int n = 0; n < N && cnt < K; ++n) {
      float dx = X[n] - qx, dy = X[N + n] - qy, dz = X[2 * N + n] - qz;
      if (dx * dx + dy * dy + dz * dz <= r2) o[cnt++] = n;
    }
    int f = (cnt > 0) ? o[0] : 0;
    for (int j = cnt; j < K; ++j) o[j] = f;
  }
}

// grouped tensor [B,3+D,K*S]: c<3 -> xyz[gi]-newxyz ; else pts[c-3][gi]
__global__ void sa_group_kernel(const float* __restrict__ xyz, const float* __restrict__ nxyz,
                                const float* __restrict__ pts, const int* __restrict__ gi,
                                float* __restrict__ out, int B, int N, int S, int K, int D)
{
  const int Cin = 3 + D;
  size_t total = (size_t)B * Cin * K * S;
  GSL(i, total) {
    int s = (int)(i % S); size_t r = i / S;
    int k = (int)(r % K); r /= K;
    int c = (int)(r % Cin); int b = (int)(r / Cin);
    int g = gi[((size_t)b * S + s) * K + k];
    float v;
    if (c < 3) v = xyz[((size_t)b * 3 + c) * N + g] - nxyz[((size_t)b * 3 + c) * S + s];
    else       v = pts[((size_t)b * D + (c - 3)) * N + g];
    out[i] = v;
  }
}

__global__ void maxk_kernel(const float* __restrict__ in, float* __restrict__ out,
                            int B, int C, int K, int S)
{
  size_t total = (size_t)B * C * S;
  GSL(i, total) {
    int s = (int)(i % S); size_t r = i / S;
    int c = (int)(r % C); int b = (int)(r / C);
    const float* p = in + ((size_t)b * C + c) * (size_t)K * S + s;
    float m = p[0];
    for (int k = 1; k < K; ++k) m = fmaxf(m, p[(size_t)k * S]);
    out[i] = m;
  }
}

__global__ void knn3_kernel(const float* __restrict__ A, const float* __restrict__ Bp,
                            int* __restrict__ idx, float* __restrict__ w,
                            int B, int Na, int Nb)
{
  GSL(i, (size_t)B * Na) {
    int b = (int)(i / Na), n = (int)(i % Na);
    const float* pa = A + (size_t)b * 3 * Na;
    const float* pb = Bp + (size_t)b * 3 * Nb;
    float ax = pa[n], ay = pa[Na + n], az = pa[2 * Na + n];
    float d0 = 1e30f, d1 = 1e30f, d2 = 1e30f; int i0 = 0, i1 = 0, i2 = 0;
    for (int m = 0; m < Nb; ++m) {
      float dx = pb[m] - ax, dy = pb[Nb + m] - ay, dz = pb[2 * Nb + m] - az;
      float d = dx * dx + dy * dy + dz * dz;
      if (d < d0)      { d2 = d1; i2 = i1; d1 = d0; i1 = i0; d0 = d; i0 = m; }
      else if (d < d1) { d2 = d1; i2 = i1; d1 = d; i1 = m; }
      else if (d < d2) { d2 = d; i2 = m; }
    }
    float w0 = 1.f / (d0 + 1e-8f), w1 = 1.f / (d1 + 1e-8f), w2 = 1.f / (d2 + 1e-8f);
    float ws = w0 + w1 + w2;
    int* oi = idx + i * 3; float* ow = w + i * 3;
    oi[0] = i0; oi[1] = i1; oi[2] = i2;
    ow[0] = w0 / ws; ow[1] = w1 / ws; ow[2] = w2 / ws;
  }
}

__global__ void interp3_kernel(const float* __restrict__ P2, const int* __restrict__ idx,
                               const float* __restrict__ w, float* __restrict__ out,
                               int B, int C, int Nb, int Na, int Cout, int cOff)
{
  size_t total = (size_t)B * C * Na;
  GSL(i, total) {
    int n = (int)(i % Na); size_t r = i / Na;
    int c = (int)(r % C); int b = (int)(r / C);
    const int*   ii = idx + ((size_t)b * Na + n) * 3;
    const float* ww = w   + ((size_t)b * Na + n) * 3;
    const float* p  = P2  + ((size_t)b * C + c) * Nb;
    out[((size_t)b * Cout + cOff + c) * Na + n] =
        ww[0] * p[ii[0]] + ww[1] * p[ii[1]] + ww[2] * p[ii[2]];
  }
}

// channel-block copy with optional N-broadcast (Nsrc==1)
__global__ void copyc_kernel(const float* __restrict__ src, float* __restrict__ out,
                             int B, int C, int Nsrc, int Cout, int cOff, int Nout)
{
  size_t total = (size_t)B * C * Nout;
  GSL(i, total) {
    int n = (int)(i % Nout); size_t r = i / Nout;
    int c = (int)(r % C); int b = (int)(r / C);
    int ns = (Nsrc == 1) ? 0 : n;
    out[((size_t)b * Cout + cOff + c) * Nout + n] = src[((size_t)b * C + c) * Nsrc + ns];
  }
}

__global__ void knn16_kernel(const float* __restrict__ pos, int* __restrict__ idx, int B, int N)
{
  GSL(i, (size_t)B * N) {
    int b = (int)(i / N), n = (int)(i % N);
    const float* p = pos + (size_t)b * 3 * N;
    float ax = p[n], ay = p[N + n], az = p[2 * N + n];
    float db[16]; int ib[16];
    for (int j = 0; j < 16; ++j) { db[j] = 1e30f; ib[j] = 0; }
    for (int m = 0; m < N; ++m) {
      float dx = p[m] - ax, dy = p[N + m] - ay, dz = p[2 * N + m] - az;
      float d = dx * dx + dy * dy + dz * dz;
      if (d < db[15]) {
        int j = 15;
        while (j > 0 && db[j - 1] > d) { db[j] = db[j - 1]; ib[j] = ib[j - 1]; --j; }
        db[j] = d; ib[j] = m;
      }
    }
    int* o = idx + i * 16;
    for (int j = 0; j < 16; ++j) o[j] = ib[j];
  }
}

__global__ void gathersub_kernel(const float* __restrict__ x, const int* __restrict__ idx,
                                 float* __restrict__ out, int B, int C, int N, int K)
{
  size_t total = (size_t)B * C * N * K;
  GSL(i, total) {
    int j = (int)(i % K); size_t r = i / K;
    int n = (int)(r % N); r /= N;
    int c = (int)(r % C); int b = (int)(r / C);
    int g = idx[((size_t)b * N + n) * K + j];
    const float* xc = x + ((size_t)b * C + c) * N;
    out[i] = xc[n] - xc[g];
  }
}

// s3 = q[n] - k[idx] + pe + ur
__global__ void attnpre_kernel(const float* __restrict__ q, const float* __restrict__ k,
                               const int* __restrict__ idx, const float* __restrict__ pe,
                               const float* __restrict__ ur, float* __restrict__ out,
                               int B, int C, int N, int K)
{
  size_t total = (size_t)B * C * N * K;
  GSL(i, total) {
    int j = (int)(i % K); size_t r = i / K;
    int n = (int)(r % N); r /= N;
    int c = (int)(r % C); int b = (int)(r / C);
    int g = idx[((size_t)b * N + n) * K + j];
    size_t cc = (size_t)b * C + c;
    out[i] = q[cc * N + n] - k[cc * N + g] + pe[i] + ur[i];
  }
}

// val = v[idx] + pe + ur
__global__ void gatheradd_kernel(const float* __restrict__ v, const int* __restrict__ idx,
                                 const float* __restrict__ pe, const float* __restrict__ ur,
                                 float* __restrict__ out, int B, int C, int N, int K)
{
  size_t total = (size_t)B * C * N * K;
  GSL(i, total) {
    int j = (int)(i % K); size_t r = i / K;
    int n = (int)(r % N); r /= N;
    int c = (int)(r % C); int b = (int)(r / C);
    int g = idx[((size_t)b * N + n) * K + j];
    out[i] = v[((size_t)b * C + c) * N + g] + pe[i] + ur[i];
  }
}

__global__ void softmax_kernel(float* __restrict__ x, size_t rows, int K)
{
  GSL(i, rows) {
    float* p = x + i * K;
    float mx = p[0];
    for (int j = 1; j < K; ++j) mx = fmaxf(mx, p[j]);
    float s = 0.f;
    for (int j = 0; j < K; ++j) { float e = expf(p[j] - mx); p[j] = e; s += e; }
    float inv = 1.f / s;
    for (int j = 0; j < K; ++j) p[j] *= inv;
  }
}

__global__ void wsum_kernel(const float* __restrict__ a, const float* __restrict__ v,
                            float* __restrict__ out, size_t rows, int K)
{
  GSL(i, rows) {
    const float* pa = a + i * K;
    const float* pv = v + i * K;
    float s = 0.f;
    for (int j = 0; j < K; ++j) s += pa[j] * pv[j];
    out[i] = s;
  }
}

// out[b,c,(2n+u)*K+j] = s_u[b,c,n*K+j]
__global__ void interleave2_kernel(const float* __restrict__ s0, const float* __restrict__ s1,
                                   float* __restrict__ out, int B, int C, int N, int K)
{
  size_t total = (size_t)B * C * N * K;
  GSL(i, total) {
    int j = (int)(i % K); size_t r = i / K;
    int n = (int)(r % N); r /= N;
    int c = (int)(r % C); int b = (int)(r / C);
    size_t o0 = (((size_t)b * C + c) * (2 * (size_t)N) + 2 * n) * K + j;
    out[o0]     = s0[i];
    out[o0 + K] = s1[i];
  }
}

// out[b,c,(2n+u)*K+j] = src[b,c,n*K+j] (repeat along n; K=1 handles plain repeat)
__global__ void repeat2_kernel(const float* __restrict__ src, float* __restrict__ out,
                               int B, int C, int N, int K)
{
  size_t total = (size_t)B * C * N * K;
  GSL(i, total) {
    int j = (int)(i % K); size_t r = i / K;
    int n = (int)(r % N); r /= N;
    int c = (int)(r % C); int b = (int)(r / C);
    size_t o0 = (((size_t)b * C + c) * (2 * (size_t)N) + 2 * n) * K + j;
    float v = src[i];
    out[o0] = v; out[o0 + K] = v;
  }
}

// ConvTranspose weight [256,64,2] -> Wt[u][o][c] row-major (f32)
__global__ void wtr_kernel(const float* __restrict__ w, float* __restrict__ out)
{
  GSL(i, (size_t)2 * 64 * 256) {
    int c = (int)(i % 256); int o = (int)((i / 256) % 64); int u = (int)(i / (256 * 64));
    out[i] = w[(size_t)c * 128 + o * 2 + u];
  }
}

__global__ void relu_kernel(const float* __restrict__ in, float* __restrict__ out, size_t total)
{
  GSL(i, total) { float v = in[i]; out[i] = v > 0.f ? v : 0.f; }
}

__global__ void finalpos_kernel(const float* __restrict__ coarse, const float* __restrict__ delta,
                                float* __restrict__ out, int B, int N)
{
  size_t total = (size_t)B * 3 * 2 * N;
  GSL(i, total) {
    int m = (int)(i % (2 * N)); size_t r = i / (2 * N);
    int c = (int)(r % 3); int b = (int)(r / 3);
    out[i] = coarse[((size_t)b * 3 + c) * N + (m >> 1)] + delta[i];
  }
}

// ---------------------------------------------------------------------------
// Host-side orchestration
// ---------------------------------------------------------------------------
struct Bump {
  char* base; size_t off;
  Bump(void* b, size_t o) : base((char*)b), off(o) {}
  float* f(size_t n) { float* r = (float*)(base + off); off += ((n * 4) + 255) & ~(size_t)255; return r; }
  int*   i(size_t n) { int*   r = (int*)(base + off);   off += ((n * 4) + 255) & ~(size_t)255; return r; }
};

static inline unsigned gblocks(size_t total) {
  size_t b = (total + 255) / 256;
  return (unsigned)(b > 16384 ? 16384 : (b == 0 ? 1 : b));
}

struct WH { const _Float16* h; int Cp; };

// Convert a weight matrix to the padded f16 arena (rows padded to 64-coverage).
static WH wprep(hipStream_t st, Bump& S, const float* W, int O, int C) {
  int Cp = (C + 31) & ~31;
  int Op = ((O + 63) / 64) * 64;
  size_t nh = (size_t)Op * Cp;
  _Float16* p = (_Float16*)S.f((nh + 1) / 2);
  w2h_kernel<<<gblocks(nh), 256, 0, st>>>(W, p, O, C, Op, Cp);
  WH r; r.h = p; r.Cp = Cp; return r;
}

static void gemm(hipStream_t st, const float* X, WH w, const float* bias,
                 const float* gam, const float* bet, const float* res, float* Y,
                 int Bb, int C, int O, int Nn, int act)
{
  dim3 g((unsigned)(Nn / 16), (unsigned)((O + 63) / 64), (unsigned)Bb);
  gemm_wmma_kernel<<<g, 128, 0, st>>>(X, w.h, bias, gam, bet, res, Y, C, w.Cp, O, Nn, act);
}

// Param index layout (JAX pytree order: dict keys sorted, leaves b-before-w).
enum {
  I_COARSE = 0, I_FEAT = 1, I_SEED = 2, I_SEEDF = 3,
  I_D1B = 4, I_D1W = 5, I_D2B = 6, I_D2W = 7,
  I_FP1 = 8, I_FP2 = 16, I_FP3 = 24,   // per layer: bn.b, bn.g, conv.b, conv.w
  I_MDF = 32,                           // c1.b, c1.w, c2.b, c2.w, sc.b, sc.w
  I_SA1 = 38, I_SA2 = 50, I_SA3 = 62,   // 3 layers x 4
  I_UT1 = 74, I_UT2 = 102               // 28 leaves each, see offsets below
};
// ut offsets: +0 attn1.b +1 attn1.w +2 attn2.b +3 attn2.w +4 attn_bn.b +5 attn_bn.g
//  +6 end.b +7 end.w +8 k.b +9 k.w +10..15 mlp_v(c1b,c1w,c2b,c2w,scb,scw)
//  +16 pos1.b +17 pos1.w +18 pos2.b +19 pos2.w +20 pos_bn.b +21 pos_bn.g
//  +22 q.b +23 q.w +24 up.b +25 up.w +26 v.b +27 v.w

static void sa_run(hipStream_t st, const float* const* Pm, int pb,
                   const float* xyz, const float* pts, int Bb, int Nn, int D,
                   int S, int Kq, float radius, int c1, int c2, int c3,
                   float* oxyz, float* ofeat, Bump Sc)
{
  int* fi = Sc.i((size_t)Bb * S);
  fps_kernel<<<Bb, 256, 0, st>>>(xyz, Nn, S, fi);
  gatherp_kernel<<<gblocks((size_t)Bb * 3 * S), 256, 0, st>>>(xyz, fi, oxyz, Bb, 3, Nn, S);
  int* gi = Sc.i((size_t)Bb * S * Kq);
  qball_kernel<<<gblocks((size_t)Bb * S), 256, 0, st>>>(xyz, oxyz, gi, radius * radius, Bb, Nn, S, Kq);
  const int Cin = 3 + D;
  const int NF  = Kq * S;
  float* g0 = Sc.f((size_t)Bb * Cin * NF);
  sa_group_kernel<<<gblocks((size_t)Bb * Cin * NF), 256, 0, st>>>(xyz, oxyz, pts, gi, g0, Bb, Nn, S, Kq, D);
  float* b1 = Sc.f((size_t)Bb * c1 * NF);
  gemm(st, g0, wprep(st, Sc, Pm[pb + 3], c1, Cin), Pm[pb + 2], Pm[pb + 1], Pm[pb + 0], 0, b1, Bb, Cin, c1, NF, ACT_RELU);
  float* b2 = Sc.f((size_t)Bb * c2 * NF);
  gemm(st, b1, wprep(st, Sc, Pm[pb + 7], c2, c1), Pm[pb + 6], Pm[pb + 5], Pm[pb + 4], 0, b2, Bb, c1, c2, NF, ACT_RELU);
  float* b3 = Sc.f((size_t)Bb * c3 * NF);
  gemm(st, b2, wprep(st, Sc, Pm[pb + 11], c3, c2), Pm[pb + 10], Pm[pb + 9], Pm[pb + 8], 0, b3, Bb, c2, c3, NF, ACT_RELU);
  maxk_kernel<<<gblocks((size_t)Bb * c3 * S), 256, 0, st>>>(b3, ofeat, Bb, c3, Kq, S);
}

static void uptrans_run(hipStream_t st, const float* const* Pm, int ub,
                        const float* pos, const float* keyf, const float* qryf,
                        const float* upfeat, int Bb, int n, int up, float* outY, Bump S)
{
  const int KN = 16;
  const size_t nk = (size_t)n * KN;

  float* cat = S.f((size_t)Bb * 256 * n);
  copyc_kernel<<<gblocks((size_t)Bb * 128 * n), 256, 0, st>>>(keyf, cat, Bb, 128, n, 256, 0, n);
  copyc_kernel<<<gblocks((size_t)Bb * 128 * n), 256, 0, st>>>(qryf, cat, Bb, 128, n, 256, 128, n);
  float* t1  = S.f((size_t)Bb * 128 * n);
  gemm(st, cat, wprep(st, S, Pm[ub + 11], 128, 256), Pm[ub + 10], 0, 0, 0, t1, Bb, 256, 128, n, ACT_RELU);
  float* scv = S.f((size_t)Bb * 128 * n);
  gemm(st, cat, wprep(st, S, Pm[ub + 15], 128, 256), Pm[ub + 14], 0, 0, 0, scv, Bb, 256, 128, n, ACT_NONE);
  float* value = S.f((size_t)Bb * 128 * n);
  gemm(st, t1, wprep(st, S, Pm[ub + 13], 128, 128), Pm[ub + 12], 0, 0, scv, value, Bb, 128, 128, n, ACT_NONE);

  float* kf = S.f((size_t)Bb * 64 * n);
  gemm(st, keyf,  wprep(st, S, Pm[ub + 9], 64, 128),  Pm[ub + 8],  0, 0, 0, kf, Bb, 128, 64, n, ACT_NONE);
  float* qf = S.f((size_t)Bb * 64 * n);
  gemm(st, qryf,  wprep(st, S, Pm[ub + 23], 64, 128), Pm[ub + 22], 0, 0, 0, qf, Bb, 128, 64, n, ACT_NONE);
  float* vf = S.f((size_t)Bb * 64 * n);
  gemm(st, value, wprep(st, S, Pm[ub + 27], 64, 128), Pm[ub + 26], 0, 0, 0, vf, Bb, 128, 64, n, ACT_NONE);
  float* uf = S.f((size_t)Bb * 64 * n);
  gemm(st, upfeat, wprep(st, S, Pm[ub + 25], 64, 128), Pm[ub + 24], 0, 0, 0, uf, Bb, 128, 64, n, ACT_NONE);

  int* idx = S.i((size_t)Bb * n * KN);
  knn16_kernel<<<gblocks((size_t)Bb * n), 256, 0, st>>>(pos, idx, Bb, n);

  float* posrel = S.f((size_t)Bb * 3 * nk);
  gathersub_kernel<<<gblocks((size_t)Bb * 3 * nk), 256, 0, st>>>(pos, idx, posrel, Bb, 3, n, KN);
  float* pet = S.f((size_t)Bb * 64 * nk);
  gemm(st, posrel, wprep(st, S, Pm[ub + 17], 64, 3), Pm[ub + 16], Pm[ub + 21], Pm[ub + 20], 0, pet, Bb, 3, 64, (int)nk, ACT_RELU);
  float* pe = S.f((size_t)Bb * 64 * nk);
  gemm(st, pet, wprep(st, S, Pm[ub + 19], 64, 64), Pm[ub + 18], 0, 0, 0, pe, Bb, 64, 64, (int)nk, ACT_NONE);
  float* ur = S.f((size_t)Bb * 64 * nk);
  gathersub_kernel<<<gblocks((size_t)Bb * 64 * nk), 256, 0, st>>>(uf, idx, ur, Bb, 64, n, KN);
  float* s3 = S.f((size_t)Bb * 64 * nk);
  attnpre_kernel<<<gblocks((size_t)Bb * 64 * nk), 256, 0, st>>>(qf, kf, idx, pe, ur, s3, Bb, 64, n, KN);
  float* h = S.f((size_t)Bb * 256 * nk);
  gemm(st, s3, wprep(st, S, Pm[ub + 1], 256, 64), Pm[ub + 0], Pm[ub + 5], Pm[ub + 4], 0, h, Bb, 64, 256, (int)nk, ACT_RELU);

  float* attn; int nout;
  if (up == 0) {
    attn = s3;  // s3 is dead now
    gemm(st, h, wprep(st, S, Pm[ub + 3], 64, 256), Pm[ub + 2], 0, 0, 0, attn, Bb, 256, 64, (int)nk, ACT_NONE);
    nout = n;
  } else {
    float* Wt = S.f((size_t)2 * 64 * 256);
    wtr_kernel<<<gblocks((size_t)2 * 64 * 256), 256, 0, st>>>(Pm[ub + 3], Wt);
    float* a0 = s3;
    float* a1 = S.f((size_t)Bb * 64 * nk);
    gemm(st, h, wprep(st, S, Wt, 64, 256),            Pm[ub + 2], 0, 0, 0, a0, Bb, 256, 64, (int)nk, ACT_NONE);
    gemm(st, h, wprep(st, S, Wt + 64 * 256, 64, 256), Pm[ub + 2], 0, 0, 0, a1, Bb, 256, 64, (int)nk, ACT_NONE);
    attn = h;  // reuse h's first half [Bb,64,2*nk]
    interleave2_kernel<<<gblocks((size_t)Bb * 64 * nk), 256, 0, st>>>(a0, a1, attn, Bb, 64, n, KN);
    nout = 2 * n;
  }
  softmax_kernel<<<gblocks((size_t)Bb * 64 * nout), 256, 0, st>>>(attn, (size_t)Bb * 64 * nout, KN);

  float* val = pet;  // pet is dead now
  gatheradd_kernel<<<gblocks((size_t)Bb * 64 * nk), 256, 0, st>>>(vf, idx, pe, ur, val, Bb, 64, n, KN);
  const float* valf = val;
  if (up) {
    float* vrep = h + (size_t)Bb * 64 * 2 * nk;  // h's second half
    repeat2_kernel<<<gblocks((size_t)Bb * 64 * nk), 256, 0, st>>>(val, vrep, Bb, 64, n, KN);
    valf = vrep;
  }
  float* ws = S.f((size_t)Bb * 64 * nout);
  wsum_kernel<<<gblocks((size_t)Bb * 64 * nout), 256, 0, st>>>(attn, valf, ws, (size_t)Bb * 64 * nout, KN);

  const float* id = value;
  if (up) {
    float* idr = S.f((size_t)Bb * 128 * nout);
    repeat2_kernel<<<gblocks((size_t)Bb * 128 * n), 256, 0, st>>>(value, idr, Bb, 128, n, 1);
    id = idr;
  }
  gemm(st, ws, wprep(st, S, Pm[ub + 7], 128, 64), Pm[ub + 6], 0, 0, id, outY, Bb, 64, 128, nout, ACT_NONE);
}

extern "C" void kernel_launch(void* const* d_in, const int* in_sizes, int n_in,
                              void* d_out, int out_size, void* d_ws, size_t ws_size,
                              hipStream_t stream)
{
  (void)in_sizes; (void)out_size; (void)ws_size;
  const float* Pm[130];
  for (int i = 0; i < 130; ++i) Pm[i] = (i < n_in) ? (const float*)d_in[i] : nullptr;

  const int B = 4, N = 2048, NS = 512;
  const float* coarse = Pm[I_COARSE];
  const float* feat   = Pm[I_FEAT];
  const float* seed   = Pm[I_SEED];
  const float* seedf  = Pm[I_SEEDF];

  Bump P(d_ws, 0);
  float* featup = P.f((size_t)B * 128 * N);
  float* l1xyz  = P.f((size_t)B * 3 * 512);
  float* l1     = P.f((size_t)B * 640 * 512);
  float* l2xyz  = P.f((size_t)B * 3 * 128);
  float* l2     = P.f((size_t)B * 768 * 128);
  float* l3     = P.f((size_t)B * 1024);
  float* l2n    = P.f((size_t)B * 768 * 128);
  float* l1n    = P.f((size_t)B * 640 * 512);
  float* l0     = P.f((size_t)B * 128 * N);
  float* Hbuf   = P.f((size_t)B * 128 * N);
  float* fchild = P.f((size_t)B * 128 * 2 * N);
  const size_t scratch = P.off;

  float* out_pcd = (float*)d_out;
  float* out_K   = out_pcd + (size_t)B * 3 * 2 * N;

  // ---- feat_up: 3-NN interp of seed_feat onto coarse
  {
    Bump S(d_ws, scratch);
    int*   idx = S.i((size_t)B * N * 3);
    float* w   = S.f((size_t)B * N * 3);
    knn3_kernel<<<gblocks((size_t)B * N), 256, 0, stream>>>(coarse, seed, idx, w, B, N, NS);
    interp3_kernel<<<gblocks((size_t)B * 128 * N), 256, 0, stream>>>(seedf, idx, w, featup, B, 128, NS, N, 128, 0);
  }
  // ---- SA1
  {
    Bump S(d_ws, scratch);
    float* l1f = S.f((size_t)B * 128 * 512);
    sa_run(stream, Pm, I_SA1, coarse, coarse, B, N, 3, 512, 32, 0.1f, 64, 64, 128, l1xyz, l1f, S);
    copyc_kernel<<<gblocks((size_t)B * 128 * 512), 256, 0, stream>>>(l1f,  l1, B, 128, 512, 640, 0,   512);
    copyc_kernel<<<gblocks((size_t)B * 512 * 512), 256, 0, stream>>>(feat, l1, B, 512, 1,   640, 128, 512);
  }
  // ---- SA2
  {
    Bump S(d_ws, scratch);
    float* l2f = S.f((size_t)B * 256 * 128);
    sa_run(stream, Pm, I_SA2, l1xyz, l1, B, 512, 640, 128, 32, 0.2f, 128, 128, 256, l2xyz, l2f, S);
    copyc_kernel<<<gblocks((size_t)B * 256 * 128), 256, 0, stream>>>(l2f,  l2, B, 256, 128, 768, 0,   128);
    copyc_kernel<<<gblocks((size_t)B * 512 * 128), 256, 0, stream>>>(feat, l2, B, 512, 1,   768, 256, 128);
  }
  // ---- SA3 (group_all): x = [l2_xyz ; l2], GEMMs @ N=128, max over points
  {
    Bump S(d_ws, scratch);
    float* in3 = S.f((size_t)B * 771 * 128);
    copyc_kernel<<<gblocks((size_t)B * 3 * 128), 256, 0, stream>>>(l2xyz, in3, B, 3, 128, 771, 0, 128);
    copyc_kernel<<<gblocks((size_t)B * 768 * 128), 256, 0, stream>>>(l2, in3, B, 768, 128, 771, 3, 128);
    float* b1 = S.f((size_t)B * 256 * 128);
    gemm(stream, in3, wprep(stream, S, Pm[I_SA3 + 3], 256, 771), Pm[I_SA3 + 2], Pm[I_SA3 + 1], Pm[I_SA3 + 0], 0, b1, B, 771, 256, 128, ACT_RELU);
    float* b2 = S.f((size_t)B * 256 * 128);
    gemm(stream, b1, wprep(stream, S, Pm[I_SA3 + 7], 256, 256), Pm[I_SA3 + 6], Pm[I_SA3 + 5], Pm[I_SA3 + 4], 0, b2, B, 256, 256, 128, ACT_RELU);
    float* b3 = S.f((size_t)B * 512 * 128);
    gemm(stream, b2, wprep(stream, S, Pm[I_SA3 + 11], 512, 256), Pm[I_SA3 + 10], Pm[I_SA3 + 9], Pm[I_SA3 + 8], 0, b3, B, 256, 512, 128, ACT_RELU);
    float* l3f = S.f((size_t)B * 512);
    maxk_kernel<<<gblocks((size_t)B * 512), 256, 0, stream>>>(b3, l3f, B, 512, 128, 1);
    copyc_kernel<<<gblocks((size_t)B * 512), 256, 0, stream>>>(l3f,  l3, B, 512, 1, 1024, 0,   1);
    copyc_kernel<<<gblocks((size_t)B * 512), 256, 0, stream>>>(feat, l3, B, 512, 1, 1024, 512, 1);
  }
  // ---- FP3 (l3 broadcast onto l2 points)
  {
    Bump S(d_ws, scratch);
    float* fin = S.f((size_t)B * 1792 * 128);
    copyc_kernel<<<gblocks((size_t)B * 768 * 128), 256, 0, stream>>>(l2, fin, B, 768, 128, 1792, 0, 128);
    copyc_kernel<<<gblocks((size_t)B * 1024 * 128), 256, 0, stream>>>(l3, fin, B, 1024, 1, 1792, 768, 128);
    float* b1 = S.f((size_t)B * 256 * 128);
    gemm(stream, fin, wprep(stream, S, Pm[I_FP3 + 3], 256, 1792), Pm[I_FP3 + 2], Pm[I_FP3 + 1], Pm[I_FP3 + 0], 0, b1, B, 1792, 256, 128, ACT_RELU);
    float* b2 = S.f((size_t)B * 256 * 128);
    gemm(stream, b1, wprep(stream, S, Pm[I_FP3 + 7], 256, 256), Pm[I_FP3 + 6], Pm[I_FP3 + 5], Pm[I_FP3 + 4], 0, b2, B, 256, 256, 128, ACT_RELU);
    copyc_kernel<<<gblocks((size_t)B * 256 * 128), 256, 0, stream>>>(b2,   l2n, B, 256, 128, 768, 0,   128);
    copyc_kernel<<<gblocks((size_t)B * 512 * 128), 256, 0, stream>>>(feat, l2n, B, 512, 1,   768, 256, 128);
  }
  // ---- FP2
  {
    Bump S(d_ws, scratch);
    int*   idx = S.i((size_t)B * 512 * 3);
    float* w   = S.f((size_t)B * 512 * 3);
    knn3_kernel<<<gblocks((size_t)B * 512), 256, 0, stream>>>(l1xyz, l2xyz, idx, w, B, 512, 128);
    float* fin = S.f((size_t)B * 1408 * 512);
    copyc_kernel<<<gblocks((size_t)B * 640 * 512), 256, 0, stream>>>(l1, fin, B, 640, 512, 1408, 0, 512);
    interp3_kernel<<<gblocks((size_t)B * 768 * 512), 256, 0, stream>>>(l2n, idx, w, fin, B, 768, 128, 512, 1408, 640);
    float* b1 = S.f((size_t)B * 256 * 512);
    gemm(stream, fin, wprep(stream, S, Pm[I_FP2 + 3], 256, 1408), Pm[I_FP2 + 2], Pm[I_FP2 + 1], Pm[I_FP2 + 0], 0, b1, B, 1408, 256, 512, ACT_RELU);
    float* b2 = S.f((size_t)B * 128 * 512);
    gemm(stream, b1, wprep(stream, S, Pm[I_FP2 + 7], 128, 256), Pm[I_FP2 + 6], Pm[I_FP2 + 5], Pm[I_FP2 + 4], 0, b2, B, 256, 128, 512, ACT_RELU);
    copyc_kernel<<<gblocks((size_t)B * 128 * 512), 256, 0, stream>>>(b2,   l1n, B, 128, 512, 640, 0,   512);
    copyc_kernel<<<gblocks((size_t)B * 512 * 512), 256, 0, stream>>>(feat, l1n, B, 512, 1,   640, 128, 512);
  }
  // ---- FP1 -> l0
  {
    Bump S(d_ws, scratch);
    int*   idx = S.i((size_t)B * N * 3);
    float* w   = S.f((size_t)B * N * 3);
    knn3_kernel<<<gblocks((size_t)B * N), 256, 0, stream>>>(coarse, l1xyz, idx, w, B, N, 512);
    float* fin = S.f((size_t)B * 640 * N);
    interp3_kernel<<<gblocks((size_t)B * 640 * N), 256, 0, stream>>>(l1n, idx, w, fin, B, 640, 512, N, 640, 0);
    float* b1 = S.f((size_t)B * 128 * N);
    gemm(stream, fin, wprep(stream, S, Pm[I_FP1 + 3], 128, 640), Pm[I_FP1 + 2], Pm[I_FP1 + 1], Pm[I_FP1 + 0], 0, b1, B, 640, 128, N, ACT_RELU);
    gemm(stream, b1,  wprep(stream, S, Pm[I_FP1 + 7], 128, 128), Pm[I_FP1 + 6], Pm[I_FP1 + 5], Pm[I_FP1 + 4], 0, l0, B, 128, 128, N, ACT_RELU);
  }
  // ---- up-transformers
  {
    Bump S(d_ws, scratch);
    uptrans_run(stream, Pm, I_UT1, coarse, l0, l0, featup, B, N, 0, Hbuf, S);
  }
  {
    Bump S(d_ws, scratch);
    uptrans_run(stream, Pm, I_UT2, coarse, Hbuf, Hbuf, featup, B, N, 2, fchild, S);
  }
  // ---- MDF + delta heads
  {
    Bump S(d_ws, scratch);
    const int N2 = 2 * N;
    float* Hup = S.f((size_t)B * 128 * N2);
    repeat2_kernel<<<gblocks((size_t)B * 128 * N), 256, 0, stream>>>(Hbuf, Hup, B, 128, N, 1);
    float* Kcat = S.f((size_t)B * 256 * N2);
    copyc_kernel<<<gblocks((size_t)B * 128 * N2), 256, 0, stream>>>(fchild, Kcat, B, 128, N2, 256, 0,   N2);
    copyc_kernel<<<gblocks((size_t)B * 128 * N2), 256, 0, stream>>>(Hup,    Kcat, B, 128, N2, 256, 128, N2);
    float* t1 = S.f((size_t)B * 128 * N2);
    gemm(stream, Kcat, wprep(stream, S, Pm[I_MDF + 1], 128, 256), Pm[I_MDF + 0], 0, 0, 0, t1, B, 256, 128, N2, ACT_RELU);
    float* scv = S.f((size_t)B * 128 * N2);
    gemm(stream, Kcat, wprep(stream, S, Pm[I_MDF + 5], 128, 256), Pm[I_MDF + 4], 0, 0, 0, scv, B, 256, 128, N2, ACT_NONE);
    gemm(stream, t1, wprep(stream, S, Pm[I_MDF + 3], 128, 128), Pm[I_MDF + 2], 0, 0, scv, out_K, B, 128, 128, N2, ACT_NONE);  // K_curr
    float* rk = S.f((size_t)B * 128 * N2);
    relu_kernel<<<gblocks((size_t)B * 128 * N2), 256, 0, stream>>>(out_K, rk, (size_t)B * 128 * N2);
    float* d1 = S.f((size_t)B * 64 * N2);
    gemm(stream, rk, wprep(stream, S, Pm[I_D1W], 64, 128), Pm[I_D1B], 0, 0, 0, d1, B, 128, 64, N2, ACT_RELU);
    float* dl = S.f((size_t)B * 3 * N2);
    gemm(stream, d1, wprep(stream, S, Pm[I_D2W], 3, 64), Pm[I_D2B], 0, 0, 0, dl, B, 64, 3, N2, ACT_TANH);
    finalpos_kernel<<<gblocks((size_t)B * 3 * N2), 256, 0, stream>>>(coarse, dl, out_pcd, B, N);
  }
}
